// DiffLlamaFactChecker_29051158790468
// MI455X (gfx1250) — matrix operand order repaired and checked by reading, not compiled, not run on hardware
//
#include <hip/hip_runtime.h>
#include <hip/hip_bf16.h>

// ---------------------------------------------------------------------------
// Types for CDNA5 WMMA (wave32, 16x16x32 bf16 -> f32)
// ---------------------------------------------------------------------------
typedef __bf16 bf16_t;
typedef bf16_t v16bf __attribute__((ext_vector_type(16)));
typedef bf16_t v8bf  __attribute__((ext_vector_type(8)));
typedef bf16_t v4bf  __attribute__((ext_vector_type(4)));
typedef float  v8f   __attribute__((ext_vector_type(8)));

static __device__ __forceinline__ v16bf join8(v8bf lo, v8bf hi) {
  v16bf r;
#pragma unroll
  for (int i = 0; i < 8; ++i) { r[i] = lo[i]; r[8 + i] = hi[i]; }
  return r;
}

static __device__ __forceinline__ v8f wmma_bf16(v16bf a, v16bf b, v8f c) {
  return __builtin_amdgcn_wmma_f32_16x16x32_bf16(false, a, false, b, (short)0, c,
                                                 false, false);
}

// ---------------------------------------------------------------------------
// gfx1250 async global->LDS copy (ASYNCcnt-tracked, bypasses VGPRs).
// Inline asm per the CDNA5 bridge doc: portable across toolchains, avoids the
// address-space-qualified pointer types the clang builtins require.
// LDS operand = low 32 bits of the generic pointer (flat->LDS mapping uses
// addr[31:0]); global operand = 64-bit address in a VGPR pair.
// ---------------------------------------------------------------------------
static __device__ __forceinline__ void async_cp16(void* lds_dst, const void* gsrc) {
  asm volatile("global_load_async_to_lds_b128 %0, %1, off"
               :
               : "v"((unsigned)(unsigned long long)lds_dst),
                 "v"((unsigned long long)gsrc)
               : "memory");
}
static __device__ __forceinline__ void wait_async() {
  asm volatile("s_wait_asynccnt 0x0" ::: "memory");
}

// ---------------------------------------------------------------------------
// Pure-VALU cross-lane reductions via DPP (no ds_bpermute round trips).
// dpp8 permutes within 8-lane groups (xor 1/2/4); row_ror:8 swaps the two
// 8-groups of a 16-lane row ((i+8)%16 == i^8). Nothing crosses the 16-lane
// half boundary, matching the WMMA C-layout row split.
// ---------------------------------------------------------------------------
static constexpr int DPP8_XOR1 = 1 | (0 << 3) | (3 << 6) | (2 << 9) |
                                 (5 << 12) | (4 << 15) | (7 << 18) | (6 << 21);
static constexpr int DPP8_XOR2 = 2 | (3 << 3) | (0 << 6) | (1 << 9) |
                                 (6 << 12) | (7 << 15) | (4 << 18) | (5 << 21);
static constexpr int DPP8_XOR4 = 4 | (5 << 3) | (6 << 6) | (7 << 9) |
                                 (0 << 12) | (1 << 15) | (2 << 18) | (3 << 21);

template <int SEL>
static __device__ __forceinline__ float dpp8_f(float x) {
  return __builtin_bit_cast(
      float, __builtin_amdgcn_mov_dpp8(__builtin_bit_cast(int, x), SEL));
}
static __device__ __forceinline__ float dpp_ror8_f(float x) {
  // DPP16 row_ror:8 (ctrl 0x128), full row/bank masks, bound_ctrl=1
  return __builtin_bit_cast(
      float, __builtin_amdgcn_update_dpp(0, __builtin_bit_cast(int, x), 0x128,
                                         0xf, 0xf, true));
}
static __device__ __forceinline__ float row16_max(float v) {
  v = fmaxf(v, dpp8_f<DPP8_XOR1>(v));
  v = fmaxf(v, dpp8_f<DPP8_XOR2>(v));
  v = fmaxf(v, dpp8_f<DPP8_XOR4>(v));
  v = fmaxf(v, dpp_ror8_f(v));
  return v;
}
static __device__ __forceinline__ float row16_sum(float v) {
  v += dpp8_f<DPP8_XOR1>(v);
  v += dpp8_f<DPP8_XOR2>(v);
  v += dpp8_f<DPP8_XOR4>(v);
  v += dpp_ror8_f(v);
  return v;
}

// ---------------------------------------------------------------------------
// Problem constants
// ---------------------------------------------------------------------------
static constexpr int Bc    = 8;
static constexpr int Sc    = 1024;
static constexpr int Hc    = 1024;
static constexpr int HEADS = 8;
static constexpr int HD    = 128;
static constexpr float SCALE       = 0.08838834764831845f;  // 1/sqrt(128)
static constexpr float LAMBDA_INIT = 0.8f;
static constexpr float LN_EPS      = 1e-5f;

// ---------------------------------------------------------------------------
// Kernel 1: f32 -> bf16 elementwise convert (vector x4)
// ---------------------------------------------------------------------------
__global__ void cvt_f32_bf16(const float* __restrict__ in, bf16_t* __restrict__ out) {
  int i = (blockIdx.x * blockDim.x + threadIdx.x) * 4;
  float4 v = *(const float4*)(in + i);
  v4bf o;
  o[0] = (bf16_t)v.x; o[1] = (bf16_t)v.y; o[2] = (bf16_t)v.z; o[3] = (bf16_t)v.w;
  *(v4bf*)(out + i) = o;
}

// ---------------------------------------------------------------------------
// Kernel 2: transpose + convert weight  W[k][n] f32 -> Wt[n][k] bf16
// ---------------------------------------------------------------------------
__global__ void transpose_cvt(const float* __restrict__ W, bf16_t* __restrict__ Wt,
                              int dim) {
  __shared__ float tile[32][33];
  const int bx = blockIdx.x * 32;  // n tile
  const int by = blockIdx.y * 32;  // k tile
  const int tx = threadIdx.x & 31;
  const int ty = threadIdx.x >> 5;
#pragma unroll
  for (int i = ty; i < 32; i += 8)
    tile[i][tx] = W[(size_t)(by + i) * dim + bx + tx];
  __syncthreads();
#pragma unroll
  for (int i = ty; i < 32; i += 8)
    Wt[(size_t)(bx + i) * dim + by + tx] = (bf16_t)tile[tx][i];
}

// ---------------------------------------------------------------------------
// Kernel 3: lambda gate -> single float in workspace
// ---------------------------------------------------------------------------
__global__ void lambda_kernel(const float* __restrict__ lq1, const float* __restrict__ lk1,
                              const float* __restrict__ lq2, const float* __restrict__ lk2,
                              float* __restrict__ lamp) {
  __shared__ float b1[HD], b2[HD];
  const int t = threadIdx.x;
  b1[t] = lq1[t] * lk1[t];
  b2[t] = lq2[t] * lk2[t];
  __syncthreads();
  if (t == 0) {
    float s1 = 0.f, s2 = 0.f;
    for (int i = 0; i < HD; ++i) { s1 += b1[i]; s2 += b2[i]; }
    lamp[0] = __expf(s1) - __expf(s2) + LAMBDA_INIT;
  }
}

// ---------------------------------------------------------------------------
// Kernel 4: WMMA GEMM  C[M][N] = (A[M][K] @ Wt[N][K]^T + bias) * out_scale
// Block: 256 threads (8 waves), tile 128(M) x 64(N), K-step 32.
// Double-buffered async global->LDS staging overlapped with WMMAs.
// ---------------------------------------------------------------------------
template <typename OutT>
__global__ void __launch_bounds__(256, 1)
gemm_bf16_wmma(const bf16_t* __restrict__ A, const bf16_t* __restrict__ Bt,
               const float* __restrict__ bias, OutT* __restrict__ C,
               int M, int N, int K, float out_scale) {
  __shared__ bf16_t sA[2][128][40];
  __shared__ bf16_t sB[2][64][40];

  const int tid  = threadIdx.x;
  const int lane = tid & 31;
  const int wave = tid >> 5;
  const int wm   = (wave & 3) * 32;
  const int wn   = (wave >> 2) * 32;
  const int m0   = blockIdx.x * 128;
  const int n0   = blockIdx.y * 64;
  const int lr   = lane & 15;
  const int hi   = lane >> 4;
  const int r    = tid >> 2;       // 0..63
  const int cg   = (tid & 3) * 8;  // 0,8,16,24

  v8f acc[2][2] = {};

  auto stage = [&](int k0, int bf) {
    async_cp16(&sA[bf][r][cg],      &A[(size_t)(m0 + r) * K + k0 + cg]);
    async_cp16(&sA[bf][r + 64][cg], &A[(size_t)(m0 + r + 64) * K + k0 + cg]);
    async_cp16(&sB[bf][r][cg],      &Bt[(size_t)(n0 + r) * K + k0 + cg]);
  };
  stage(0, 0);
  wait_async();
  __syncthreads();

  int buf = 0;
  for (int k0 = 0; k0 < K; k0 += 32) {
    if (k0 + 32 < K) stage(k0 + 32, buf ^ 1);  // overlap with WMMAs below

    v16bf af[2], bfg[2];
#pragma unroll
    for (int i = 0; i < 2; ++i) {
      const bf16_t* ap = &sA[buf][wm + i * 16 + lr][hi * 8];
      af[i] = join8(*(const v8bf*)ap, *(const v8bf*)(ap + 16));
      const bf16_t* bp = &sB[buf][wn + i * 16 + lr][hi * 8];
      bfg[i] = join8(*(const v8bf*)bp, *(const v8bf*)(bp + 16));
    }
#pragma unroll
    for (int i = 0; i < 2; ++i)
#pragma unroll
      for (int j = 0; j < 2; ++j) acc[i][j] = wmma_bf16(af[i], bfg[j], acc[i][j]);

    wait_async();
    __syncthreads();
    buf ^= 1;
  }

#pragma unroll
  for (int i = 0; i < 2; ++i)
#pragma unroll
    for (int j = 0; j < 2; ++j) {
      const int col = n0 + wn + j * 16 + lr;
      const float bb = bias[col];
#pragma unroll
      for (int e = 0; e < 8; ++e) {
        const int row = m0 + wm + i * 16 + e + hi * 8;
        C[(size_t)row * N + col] = (OutT)((acc[i][j][e] + bb) * out_scale);
      }
    }
}

// ---------------------------------------------------------------------------
// Kernel 5: combined value path  Vct[b][hh][d][s] = V[b][s][hh*HD+d]
//                                                  - lam*V[b][s][(hh+4)*HD+d]
// Pre-transposed so attention B-fragment staging is contiguous b128 copies.
// ---------------------------------------------------------------------------
__global__ void combine_v(const bf16_t* __restrict__ V, const float* __restrict__ lamp,
                          bf16_t* __restrict__ Vct) {
  __shared__ float tile[32][33];
  const float lam = lamp[0];
  const int hh = blockIdx.y & 3;
  const int b  = blockIdx.y >> 2;
  const int d0 = (blockIdx.x & 3) * 32;   // 128/32 = 4 d tiles
  const int s0 = (blockIdx.x >> 2) * 32;  // 1024/32 = 32 s tiles
  const int tx = threadIdx.x & 31;
  const int ty = threadIdx.x >> 5;
#pragma unroll
  for (int i = ty; i < 32; i += 8) {
    const bf16_t* p = V + ((size_t)b * Sc + s0 + i) * Hc + hh * HD + d0 + tx;
    tile[i][tx] = (float)p[0] - lam * (float)p[4 * HD];
  }
  __syncthreads();
#pragma unroll
  for (int i = ty; i < 32; i += 8)
    Vct[(((size_t)b * 4 + hh) * HD + d0 + i) * Sc + s0 + tx] = (bf16_t)tile[tx][i];
}

// ---------------------------------------------------------------------------
// Kernel 6: differential flash attention (single fused value path).
// Grid: (S/128, HEADS, B). Block 256 threads = 8 waves; wave owns 16 q rows,
// streams 32 keys/step: 8 score WMMA + 8 PV WMMA, DPP-based online softmax,
// double-buffered async global->LDS staging for K and Vc tiles.
// ---------------------------------------------------------------------------
__global__ void __launch_bounds__(256, 1)
diff_attn(const bf16_t* __restrict__ Q, const bf16_t* __restrict__ Kb,
          const bf16_t* __restrict__ Vct, float* __restrict__ AO) {
  __shared__ bf16_t sK[2][32][HD + 8];  // [key][d]
  __shared__ bf16_t sV[2][HD][40];      // [d][key] (pre-transposed in Vct)
  __shared__ bf16_t sP[8][16][32];      // per-wave P scratch

  const int h     = blockIdx.y;
  const int b     = blockIdx.z;
  const int qbase = blockIdx.x * 128;
  const int tid   = threadIdx.x;
  const int lane  = tid & 31;
  const int wave  = tid >> 5;
  const int lr    = lane & 15;
  const int hi    = lane >> 4;

  // preload Q fragments (Q already carries the 1/sqrt(hd) scale)
  const int qrow = qbase + wave * 16 + lr;
  const bf16_t* qptr = Q + ((size_t)b * Sc + qrow) * Hc + h * HD;
  v16bf qf[4];
#pragma unroll
  for (int dc = 0; dc < 4; ++dc) {
    const bf16_t* p = qptr + dc * 32 + hi * 8;
    qf[dc] = join8(*(const v8bf*)p, *(const v8bf*)(p + 16));
  }

  v8f o[8] = {};
  float m[8], l[8];
#pragma unroll
  for (int e = 0; e < 8; ++e) { m[e] = -1e30f; l[e] = 0.f; }

  // staging assignments
  const int key = tid >> 3;        // K: key row 0..31
  const int dg  = (tid & 7) * 16;  // K: d group
  const int dr  = tid >> 1;        // V: d row 0..127
  const int kg  = (tid & 1) * 16;  // V: key group
  const bf16_t* kbase = Kb + (size_t)b * Sc * Hc + h * HD;
  const bf16_t* vbase = Vct + (((size_t)b * 4 + (h & 3)) * HD) * Sc;

  auto stage = [&](int k0, int bf) {
    const bf16_t* kp = kbase + (size_t)(k0 + key) * Hc + dg;
    async_cp16(&sK[bf][key][dg],     kp);
    async_cp16(&sK[bf][key][dg + 8], kp + 8);
    const bf16_t* vp = vbase + (size_t)dr * Sc + k0 + kg;
    async_cp16(&sV[bf][dr][kg],      vp);
    async_cp16(&sV[bf][dr][kg + 8],  vp + 8);
  };
  stage(0, 0);
  wait_async();
  __syncthreads();

  int buf = 0;
  for (int k0 = 0; k0 < Sc; k0 += 32) {
    if (k0 + 32 < Sc) stage(k0 + 32, buf ^ 1);  // overlap with compute

    // ---- scores: two 16-key column groups, K-dim = head dim (4 x 32) ----
    v8f s[2] = {};
#pragma unroll
    for (int cgp = 0; cgp < 2; ++cgp)
#pragma unroll
      for (int dc = 0; dc < 4; ++dc) {
        const bf16_t* bp = &sK[buf][cgp * 16 + lr][dc * 32 + hi * 8];
        v16bf bfg = join8(*(const v8bf*)bp, *(const v8bf*)(bp + 16));
        s[cgp] = wmma_bf16(qf[dc], bfg, s[cgp]);
      }

    // ---- online softmax over the 32 keys (pure-VALU DPP reductions) ----
    float p0[8], p1[8];
#pragma unroll
    for (int e = 0; e < 8; ++e) {
      const float a0 = s[0][e], a1 = s[1][e];
      const float mx = row16_max(fmaxf(a0, a1));
      const float mn = fmaxf(m[e], mx);
      const float corr = __expf(m[e] - mn);
      p0[e] = __expf(a0 - mn);
      p1[e] = __expf(a1 - mn);
      const float rs = row16_sum(p0[e] + p1[e]);
      l[e] = l[e] * corr + rs;
      m[e] = mn;
#pragma unroll
      for (int f = 0; f < 8; ++f) o[f][e] *= corr;
    }

    // ---- C-layout -> A-layout via per-wave LDS round trip ----
#pragma unroll
    for (int e = 0; e < 8; ++e) {
      sP[wave][hi * 8 + e][lr]      = (bf16_t)p0[e];
      sP[wave][hi * 8 + e][16 + lr] = (bf16_t)p1[e];
    }
    const bf16_t* pp = &sP[wave][lr][hi * 8];
    v16bf pf = join8(*(const v8bf*)pp, *(const v8bf*)(pp + 16));

    // ---- accumulate fused value path: O += P @ Vc  (8 d-column frags) ----
#pragma unroll
    for (int f = 0; f < 8; ++f) {
      const bf16_t* bv = &sV[buf][f * 16 + lr][hi * 8];
      v16bf vf = join8(*(const v8bf*)bv, *(const v8bf*)(bv + 16));
      o[f] = wmma_bf16(pf, vf, o[f]);
    }

    wait_async();
    __syncthreads();
    buf ^= 1;
  }

  // ---- epilogue: O / l, merge heads into AO[b][s][h*HD+d] ----
  float inv[8];
#pragma unroll
  for (int e = 0; e < 8; ++e) inv[e] = 1.0f / l[e];
#pragma unroll
  for (int f = 0; f < 8; ++f)
#pragma unroll
    for (int e = 0; e < 8; ++e) {
      const int row = qbase + wave * 16 + e + hi * 8;
      const int col = f * 16 + lr;
      AO[((size_t)b * Sc + row) * Hc + h * HD + col] = o[f][e] * inv[e];
    }
}

// ---------------------------------------------------------------------------
// Kernel 7: LayerNorm over hidden dim, scaled by (1 - lambda_init) -> bf16
// ---------------------------------------------------------------------------
__global__ void layernorm_bf16(const float* __restrict__ AO,
                               const float* __restrict__ ln_w,
                               const float* __restrict__ ln_b,
                               bf16_t* __restrict__ Nb) {
  __shared__ float red[34];
  const int row = blockIdx.x;
  const float* x = AO + (size_t)row * Hc;
  float s = 0.f, s2 = 0.f;
  for (int i = threadIdx.x; i < Hc; i += blockDim.x) {
    float v = x[i]; s += v; s2 += v * v;
  }
  s  = row16_sum(s);
  s2 = row16_sum(s2);
  const int wave = threadIdx.x >> 5, lane = threadIdx.x & 31;
  if ((lane & 15) == 0) {
    const int slot = wave * 2 + (lane >> 4);
    red[slot]      = s;
    red[16 + slot] = s2;
  }
  __syncthreads();
  if (threadIdx.x == 0) {
    float ts = 0.f, t2 = 0.f;
    for (int w = 0; w < 16; ++w) { ts += red[w]; t2 += red[16 + w]; }
    red[32] = ts; red[33] = t2;
  }
  __syncthreads();
  const float mu   = red[32] * (1.0f / Hc);
  const float var  = red[33] * (1.0f / Hc) - mu * mu;
  const float rstd = rsqrtf(var + LN_EPS);
  for (int i = threadIdx.x; i < Hc; i += blockDim.x) {
    float nv = (x[i] - mu) * rstd * ln_w[i] + ln_b[i];
    Nb[(size_t)row * Hc + i] = (bf16_t)(0.2f * nv);
  }
}

// ---------------------------------------------------------------------------
// Launcher
// ---------------------------------------------------------------------------
extern "C" void kernel_launch(void* const* d_in, const int* in_sizes, int n_in,
                              void* d_out, int out_size, void* d_ws, size_t ws_size,
                              hipStream_t stream) {
  (void)in_sizes; (void)n_in; (void)out_size; (void)ws_size;
  const float* hs  = (const float*)d_in[0];
  const float* Wq  = (const float*)d_in[1];
  const float* bq  = (const float*)d_in[2];
  const float* Wk  = (const float*)d_in[3];
  const float* bk  = (const float*)d_in[4];
  const float* Wv  = (const float*)d_in[5];
  const float* bv  = (const float*)d_in[6];
  const float* Wo  = (const float*)d_in[7];
  const float* bo  = (const float*)d_in[8];
  const float* lq1 = (const float*)d_in[9];
  const float* lk1 = (const float*)d_in[10];
  const float* lq2 = (const float*)d_in[11];
  const float* lk2 = (const float*)d_in[12];
  const float* lnw = (const float*)d_in[13];
  const float* lnb = (const float*)d_in[14];
  float* out = (float*)d_out;

  const size_t M = (size_t)Bc * Sc;  // 8192 rows
  char* ws = (char*)d_ws;
  size_t off = 0;
  auto take = [&](size_t bytes) {
    void* p = ws + off;
    off += (bytes + 255) & ~(size_t)255;
    return p;
  };
  bf16_t* Xb  = (bf16_t*)take(M * Hc * 2);
  bf16_t* Wtq = (bf16_t*)take((size_t)Hc * Hc * 2);
  bf16_t* Wtk = (bf16_t*)take((size_t)Hc * Hc * 2);
  bf16_t* Wtv = (bf16_t*)take((size_t)Hc * Hc * 2);
  bf16_t* Wto = (bf16_t*)take((size_t)Hc * Hc * 2);
  bf16_t* Qb  = (bf16_t*)take(M * Hc * 2);
  bf16_t* Kbf = (bf16_t*)take(M * Hc * 2);
  bf16_t* Vbf = (bf16_t*)take(M * Hc * 2);
  bf16_t* Vct = (bf16_t*)take(M * (Hc / 2) * 2);  // [B][4][HD][S]
  float*  AO  = (float*)take(M * Hc * 4);
  bf16_t* Nb  = (bf16_t*)take(M * Hc * 2);
  float*  lam = (float*)take(256);

  // 1) convert hidden states to bf16; compute lambda
  cvt_f32_bf16<<<dim3((unsigned)(M * Hc / 1024)), dim3(256), 0, stream>>>(hs, Xb);
  lambda_kernel<<<dim3(1), dim3(HD), 0, stream>>>(lq1, lk1, lq2, lk2, lam);
  // 2) transpose+convert weights
  {
    dim3 g(32, 32), blk(256);
    transpose_cvt<<<g, blk, 0, stream>>>(Wq, Wtq, Hc);
    transpose_cvt<<<g, blk, 0, stream>>>(Wk, Wtk, Hc);
    transpose_cvt<<<g, blk, 0, stream>>>(Wv, Wtv, Hc);
    transpose_cvt<<<g, blk, 0, stream>>>(Wo, Wto, Hc);
  }
  // 3) QKV projections (bf16 out); Q pre-scaled by 1/sqrt(hd)
  {
    dim3 g((unsigned)(M / 128), Hc / 64), blk(256);
    gemm_bf16_wmma<bf16_t><<<g, blk, 0, stream>>>(Xb, Wtq, bq, Qb, (int)M, Hc, Hc, SCALE);
    gemm_bf16_wmma<bf16_t><<<g, blk, 0, stream>>>(Xb, Wtk, bk, Kbf, (int)M, Hc, Hc, 1.0f);
    gemm_bf16_wmma<bf16_t><<<g, blk, 0, stream>>>(Xb, Wtv, bv, Vbf, (int)M, Hc, Hc, 1.0f);
  }
  // 4) fused value path Vc = v1 - lam*v2, pre-transposed
  combine_v<<<dim3(128, Bc * 4), dim3(256), 0, stream>>>(Vbf, lam, Vct);
  // 5) differential flash attention
  diff_attn<<<dim3(Sc / 128, HEADS, Bc), dim3(256), 0, stream>>>(Qb, Kbf, Vct, AO);
  // 6) LayerNorm -> bf16
  layernorm_bf16<<<dim3((unsigned)M), dim3(256), 0, stream>>>(AO, lnw, lnb, Nb);
  // 7) output projection (f32 out)
  gemm_bf16_wmma<float><<<dim3((unsigned)(M / 128), Hc / 64), dim3(256), 0, stream>>>(
      Nb, Wto, bo, out, (int)M, Hc, Hc, 1.0f);
}
